// RSSM_37434934952048
// MI455X (gfx1250) — compile-verified, matrix operand order checked by reading, not run on hardware
//
#include <hip/hip_runtime.h>

typedef __bf16 bf16;
typedef bf16  v16bf __attribute__((ext_vector_type(16)));
typedef bf16  v8bf  __attribute__((ext_vector_type(8)));
typedef bf16  v4bf  __attribute__((ext_vector_type(4)));
typedef float v8f   __attribute__((ext_vector_type(8)));

#define B_    2048
#define T_    64
#define ADIM  6
#define STOCH 30
#define DETER 200
#define EMBED 1024
#define OUTC  380

#define HSTR  224   // hid/deter row stride (bf16); 448B rows, 16B aligned
#define GSTR  208   // gate buffer row stride
#define ESTR  512   // embed half-row stride (bf16)

struct Smem {
  bf16  x_in[16 * 64];     // [stoch(30) | action(6) | zero pad]
  bf16  hid [16 * HSTR];   // x / h_pr / h_po (reused)
  bf16  det [16 * HSTR];   // deter state
  bf16  rbuf[16 * GSTR];   // r gate, then n
  bf16  zbuf[16 * GSTR];   // z gate
  float xnb [16 * GSTR];   // xn (f32)
  float pq  [16 * 64];     // p / q head staging
  bf16  embh[16 * ESTR];   // half of embed tile, bf16
};
static_assert(sizeof(Smem) <= 65536, "LDS over 64KB");

__device__ __forceinline__ float sigm(float x)  { return 1.f / (1.f + __expf(-x)); }
__device__ __forceinline__ float eluf(float x)  { return x > 0.f ? x : __expf(x) - 1.f; }
__device__ __forceinline__ float softp(float x) { return fmaxf(x, 0.f) + log1pf(__expf(-fabsf(x))); }

// A-fragment (16x32 bf16) from LDS row-major buffer; ISA 16-bit A layout:
// lane<16 -> K {kc*32+0..7, kc*32+16..23}; lanes 16..31 -> +8.
__device__ __forceinline__ v16bf afrag_lds(const bf16* buf, int kstr, int kc, int m, int kh) {
  const bf16* p = buf + m * kstr + kc * 32 + kh * 8;
  v8bf lo = *(const v8bf*)p;
  v8bf hi = *(const v8bf*)(p + 16);
  return __builtin_shufflevector(lo, hi, 0,1,2,3,4,5,6,7,8,9,10,11,12,13,14,15);
}

// B-fragment from packed weights: fragment = 32 lanes x 16 bf16 (1KB), lane-contiguous.
// zoff is a laundered (opaque) zero so the load address is loop-variant: defeats
// LICM across the 64-step recurrence while keeping the base pointer's global
// (addrspace 1) provenance -> global_load_b128, not flat_load.
__device__ __forceinline__ v16bf bfrag(const bf16* wp, size_t zoff, int fidx, int lane) {
  return *(const v16bf*)(wp + zoff + ((size_t)fidx << 9) + (lane << 4));
}

__device__ __forceinline__ v8f wmma_bf(v16bf a, v16bf b, v8f c) {
  return __builtin_amdgcn_wmma_f32_16x16x32_bf16(false, a, false, b, (short)0, c, false, false);
}

template<int NT>
__device__ __forceinline__ void init_acc(v8f* acc, const float* b, int m, int ntOff) {
#pragma unroll
  for (int nt = 0; nt < NT; ++nt) {
    float bv = b[(ntOff + nt) * 16 + m];
    v8f a;
#pragma unroll
    for (int v = 0; v < 8; ++v) a[v] = bv;
    acc[nt] = a;
  }
}

template<int NT>
__device__ __forceinline__ void gemm_k(v8f* acc, const bf16* abuf, int kstr, int nkc,
                                       const bf16* wp, size_t zoff, int kcTot, int kcOff,
                                       int ntOff, int m, int kh, int lane) {
  for (int kc = 0; kc < nkc; ++kc) {
    v16bf a = afrag_lds(abuf, kstr, kc, m, kh);
#pragma unroll
    for (int nt = 0; nt < NT; ++nt) {
      v16bf b = bfrag(wp, zoff, (ntOff + nt) * kcTot + kcOff + kc, lane);
      acc[nt] = wmma_bf(a, b, acc[nt]);
    }
  }
}

// ---------------- weight/bias packing ----------------

__global__ void pack_w(const float* __restrict__ W, bf16* __restrict__ dst,
                       int Klocal, int N, int ldW, int colOff, int rowOff,
                       int KCl, int NT, int kcTot, int kcOff) {
  int idx = blockIdx.x * 256 + threadIdx.x;
  int total = KCl * NT * 512;
  if (idx >= total) return;
  int f = idx >> 9, r = idx & 511;
  int lane = r >> 4, e = r & 15;
  int nt = f / KCl, kc = f - nt * KCl;
  int kh = lane >> 4;
  int k  = kc * 32 + kh * 8 + (e < 8 ? e : 16 + (e - 8));
  int n  = nt * 16 + (lane & 15);
  float v = 0.f;
  if (k < Klocal && n < N) v = W[(size_t)(rowOff + k) * ldW + colOff + n];
  dst[(((size_t)nt * kcTot + kcOff + kc) << 9) + r] = (bf16)v;
}

__global__ void pack_b(const float* __restrict__ b1, const float* __restrict__ b2,
                       float* __restrict__ dst, int n, int off1, int off2, int pad) {
  int i = blockIdx.x * 256 + threadIdx.x;
  if (i >= pad) return;
  float v = 0.f;
  if (i < n) { v = b1[off1 + i]; if (b2) v += b2[off2 + i]; }
  dst[i] = v;
}

// ---------------- per-wave step body ----------------
// One block = 4 waves sharing a 16-row batch tile; wave wv owns N-tiles
// [ntOff, ntOff+NT). Block-wide elementwise passes + barriers are identical in
// every instantiation, so all waves hit the same barrier sequence.

template<int NT>
__device__ __forceinline__ void run_all(
    int ntOff, int wv, int tid, Smem& sm,
    const float* __restrict__ embed, const float* __restrict__ action,
    const float* __restrict__ eps_p, const float* __restrict__ eps_q,
    const bf16* __restrict__ Win,
    const bf16* __restrict__ Wihr, const bf16* __restrict__ Wihz, const bf16* __restrict__ Wihn,
    const bf16* __restrict__ Whhr, const bf16* __restrict__ Whhz, const bf16* __restrict__ Whhn,
    const bf16* __restrict__ Wpr1, const bf16* __restrict__ Wpr2,
    const bf16* __restrict__ Wpo1, const bf16* __restrict__ Wpo2,
    const float* __restrict__ bias, float* __restrict__ out, int b0) {
  const int lane = tid & 31;
  const int m = lane & 15, kh = lane >> 4;

  const float* Bin  = bias + 0;
  const float* Br   = bias + 208;
  const float* Bz   = bias + 416;
  const float* Bxn  = bias + 624;
  const float* Bhn  = bias + 832;
  const float* Bpr1 = bias + 1040;
  const float* Bpr2 = bias + 1248;
  const float* Bpo1 = bias + 1312;
  const float* Bpo2 = bias + 1520;

  v8f acc[NT];
  v8f acc1[1];

  for (int t = 0; t < T_; ++t) {
    // Opaque zero added to every weight-fragment address: loop-variant address
    // defeats LICM (no giant hoist/spill), base pointers keep global provenance.
    size_t zoff = 0;
    asm volatile("" : "+s"(zoff));

    // ---- inject action (cols 30..35); stoch cols 0..29 carried from prev step ----
    if (tid < 16 * ADIM) {
      int row = tid / ADIM, col = tid - row * ADIM;
      sm.x_in[row * 64 + 30 + col] =
          (bf16)action[((size_t)(b0 + row) * T_ + t) * ADIM + col];
    }
    __syncthreads();

    // ---- x = ELU([stoch|a] @ W_in + b_in) -> hid ----
    init_acc<NT>(acc, Bin, m, ntOff);
    gemm_k<NT>(acc, sm.x_in, 64, 2, Win, zoff, 2, 0, ntOff, m, kh, lane);
#pragma unroll
    for (int nt = 0; nt < NT; ++nt)
#pragma unroll
      for (int v = 0; v < 8; ++v)
        sm.hid[(v + 8 * kh) * HSTR + (ntOff + nt) * 16 + m] = (bf16)eluf(acc[nt][v]);
    __syncthreads();

    // ---- r = sigmoid(x@Wih_r + h@Whh_r + b) ----
    init_acc<NT>(acc, Br, m, ntOff);
    gemm_k<NT>(acc, sm.hid, HSTR, 7, Wihr, zoff, 7, 0, ntOff, m, kh, lane);
    gemm_k<NT>(acc, sm.det, HSTR, 7, Whhr, zoff, 7, 0, ntOff, m, kh, lane);
#pragma unroll
    for (int nt = 0; nt < NT; ++nt)
#pragma unroll
      for (int v = 0; v < 8; ++v)
        sm.rbuf[(v + 8 * kh) * GSTR + (ntOff + nt) * 16 + m] = (bf16)sigm(acc[nt][v]);

    // ---- z ----
    init_acc<NT>(acc, Bz, m, ntOff);
    gemm_k<NT>(acc, sm.hid, HSTR, 7, Wihz, zoff, 7, 0, ntOff, m, kh, lane);
    gemm_k<NT>(acc, sm.det, HSTR, 7, Whhz, zoff, 7, 0, ntOff, m, kh, lane);
#pragma unroll
    for (int nt = 0; nt < NT; ++nt)
#pragma unroll
      for (int v = 0; v < 8; ++v)
        sm.zbuf[(v + 8 * kh) * GSTR + (ntOff + nt) * 16 + m] = (bf16)sigm(acc[nt][v]);

    // ---- xn = x@Wih_n + b_ih_n (f32; r scales only hn) ----
    init_acc<NT>(acc, Bxn, m, ntOff);
    gemm_k<NT>(acc, sm.hid, HSTR, 7, Wihn, zoff, 7, 0, ntOff, m, kh, lane);
#pragma unroll
    for (int nt = 0; nt < NT; ++nt)
#pragma unroll
      for (int v = 0; v < 8; ++v)
        sm.xnb[(v + 8 * kh) * GSTR + (ntOff + nt) * 16 + m] = acc[nt][v];
    __syncthreads();

    // ---- hn; n = tanh(xn + r*hn) -> rbuf ----
    init_acc<NT>(acc, Bhn, m, ntOff);
    gemm_k<NT>(acc, sm.det, HSTR, 7, Whhn, zoff, 7, 0, ntOff, m, kh, lane);
#pragma unroll
    for (int nt = 0; nt < NT; ++nt)
#pragma unroll
      for (int v = 0; v < 8; ++v) {
        int row = v + 8 * kh, col = (ntOff + nt) * 16 + m;
        float hn = acc[nt][v];
        float xn = sm.xnb[row * GSTR + col];
        float rr = (float)sm.rbuf[row * GSTR + col];
        sm.rbuf[row * GSTR + col] = (bf16)tanhf(xn + rr * hn);
      }
    __syncthreads();

    // ---- deter = (1-z)*n + z*deter (block-wide); emit deter to out ----
    for (int j = tid; j < 16 * GSTR; j += 128) {
      int row = j / GSTR, col = j - row * GSTR;
      float zz = (float)sm.zbuf[j];
      float nn = (float)sm.rbuf[j];
      float dd = (float)sm.det[row * HSTR + col];
      float nd = (1.f - zz) * nn + zz * dd;
      sm.det[row * HSTR + col] = (bf16)nd;
      if (col < DETER)
        out[((size_t)(b0 + row) * T_ + t) * OUTC + 180 + col] = nd;
    }
    __syncthreads();

    // ---- prior: h_pr = ELU(deter @ Wpr1 + b) -> hid ----
    init_acc<NT>(acc, Bpr1, m, ntOff);
    gemm_k<NT>(acc, sm.det, HSTR, 7, Wpr1, zoff, 7, 0, ntOff, m, kh, lane);
#pragma unroll
    for (int nt = 0; nt < NT; ++nt)
#pragma unroll
      for (int v = 0; v < 8; ++v)
        sm.hid[(v + 8 * kh) * HSTR + (ntOff + nt) * 16 + m] = (bf16)eluf(acc[nt][v]);
    __syncthreads();

    // ---- p = h_pr @ Wpr2 + b -> pq (one 16-col tile per wave) ----
    init_acc<1>(acc1, Bpr2, m, wv);
    gemm_k<1>(acc1, sm.hid, HSTR, 7, Wpr2, zoff, 7, 0, wv, m, kh, lane);
#pragma unroll
    for (int v = 0; v < 8; ++v)
      sm.pq[(v + 8 * kh) * 64 + wv * 16 + m] = acc1[0][v];
    __syncthreads();

    // ---- prior heads ----
    for (int j = tid; j < 16 * STOCH; j += 128) {
      int row = j / STOCH, col = j - row * STOCH;
      float pm = sm.pq[row * 64 + col];
      float ps = softp(sm.pq[row * 64 + 30 + col]) + 0.1f;
      size_t ob = ((size_t)(b0 + row) * T_ + t) * OUTC;
      float ep = eps_p[((size_t)(b0 + row) * T_ + t) * STOCH + col];
      out[ob + 90 + col]  = pm;
      out[ob + 120 + col] = ps;
      out[ob + 150 + col] = pm + ps * ep;
    }

    // ---- posterior hidden: ELU([deter|embed] @ Wpo1 + b) -> hid ----
    init_acc<NT>(acc, Bpo1, m, ntOff);
    gemm_k<NT>(acc, sm.det, HSTR, 7, Wpo1, zoff, 39, 0, ntOff, m, kh, lane); // deter seg
    for (int half = 0; half < 2; ++half) {
      __syncthreads();  // previous readers of embh / pq done
      // cooperative f32 -> bf16 staging, vectorized: float4 loads, 8B LDS stores
      for (int idx = tid; idx < 16 * (ESTR / 4); idx += 128) {
        int row = idx >> 7, c4 = (idx & 127) << 2;
        float4 f = *(const float4*)(embed +
            ((size_t)(b0 + row) * T_ + t) * EMBED + half * ESTR + c4);
        v4bf o;
        o[0] = (bf16)f.x; o[1] = (bf16)f.y; o[2] = (bf16)f.z; o[3] = (bf16)f.w;
        *(v4bf*)(&sm.embh[row * ESTR + c4]) = o;
      }
      if (half == 0 && t + 1 < T_ && tid < 16)
        __builtin_prefetch(embed + ((size_t)(b0 + tid) * T_ + (t + 1)) * EMBED, 0, 0);
      __syncthreads();
      for (int kc = 0; kc < 16; ++kc) {
        v16bf a = afrag_lds(sm.embh, ESTR, kc, m, kh);
#pragma unroll
        for (int nt = 0; nt < NT; ++nt) {
          v16bf b = bfrag(Wpo1, zoff, (ntOff + nt) * 39 + 7 + half * 16 + kc, lane);
          acc[nt] = wmma_bf(a, b, acc[nt]);
        }
      }
    }
#pragma unroll
    for (int nt = 0; nt < NT; ++nt)
#pragma unroll
      for (int v = 0; v < 8; ++v)
        sm.hid[(v + 8 * kh) * HSTR + (ntOff + nt) * 16 + m] = (bf16)eluf(acc[nt][v]);
    __syncthreads();

    // ---- q = h_po @ Wpo2 + b -> pq ----
    init_acc<1>(acc1, Bpo2, m, wv);
    gemm_k<1>(acc1, sm.hid, HSTR, 7, Wpo2, zoff, 7, 0, wv, m, kh, lane);
#pragma unroll
    for (int v = 0; v < 8; ++v)
      sm.pq[(v + 8 * kh) * 64 + wv * 16 + m] = acc1[0][v];
    __syncthreads();

    // ---- posterior heads + stoch carry ----
    for (int j = tid; j < 16 * STOCH; j += 128) {
      int row = j / STOCH, col = j - row * STOCH;
      float qm = sm.pq[row * 64 + col];
      float qs = softp(sm.pq[row * 64 + 30 + col]) + 0.1f;
      size_t ob = ((size_t)(b0 + row) * T_ + t) * OUTC;
      float eq = eps_q[((size_t)(b0 + row) * T_ + t) * STOCH + col];
      float qst = qm + qs * eq;
      out[ob + col]      = qm;
      out[ob + 30 + col] = qs;
      out[ob + 60 + col] = qst;
      sm.x_in[row * 64 + col] = (bf16)qst;  // carried stoch
    }
    __syncthreads();
  }
}

// ---------------- main kernel ----------------

__global__ __launch_bounds__(128) void rssm_kernel(
    const float* __restrict__ embed, const float* __restrict__ action,
    const float* __restrict__ eps_p, const float* __restrict__ eps_q,
    const bf16* __restrict__ Win,
    const bf16* __restrict__ Wihr, const bf16* __restrict__ Wihz, const bf16* __restrict__ Wihn,
    const bf16* __restrict__ Whhr, const bf16* __restrict__ Whhz, const bf16* __restrict__ Whhn,
    const bf16* __restrict__ Wpr1, const bf16* __restrict__ Wpr2,
    const bf16* __restrict__ Wpo1, const bf16* __restrict__ Wpo2,
    const float* __restrict__ bias, float* __restrict__ out) {
  __shared__ Smem sm;
  const int tid = threadIdx.x;
  const int wv = tid >> 5;
  const int b0 = blockIdx.x * 16;

  for (int i = tid; i < 16 * 64; i += 128) sm.x_in[i] = (bf16)0.f;
  for (int i = tid; i < 16 * HSTR; i += 128) { sm.hid[i] = (bf16)0.f; sm.det[i] = (bf16)0.f; }
  __syncthreads();

  if (wv == 0)
    run_all<4>(0, wv, tid, sm, embed, action, eps_p, eps_q, Win, Wihr, Wihz, Wihn,
               Whhr, Whhz, Whhn, Wpr1, Wpr2, Wpo1, Wpo2, bias, out, b0);
  else if (wv == 1)
    run_all<3>(4, wv, tid, sm, embed, action, eps_p, eps_q, Win, Wihr, Wihz, Wihn,
               Whhr, Whhz, Whhn, Wpr1, Wpr2, Wpo1, Wpo2, bias, out, b0);
  else if (wv == 2)
    run_all<3>(7, wv, tid, sm, embed, action, eps_p, eps_q, Win, Wihr, Wihz, Wihn,
               Whhr, Whhz, Whhn, Wpr1, Wpr2, Wpo1, Wpo2, bias, out, b0);
  else
    run_all<3>(10, wv, tid, sm, embed, action, eps_p, eps_q, Win, Wihr, Wihz, Wihn,
               Whhr, Whhz, Whhn, Wpr1, Wpr2, Wpo1, Wpo2, bias, out, b0);
}

// ---------------- host launch ----------------

extern "C" void kernel_launch(void* const* d_in, const int* in_sizes, int n_in,
                              void* d_out, int out_size, void* d_ws, size_t ws_size,
                              hipStream_t stream) {
  const float* embed = (const float*)d_in[0];
  const float* action = (const float*)d_in[1];
  const float* np_ = (const float*)d_in[2];
  const float* nq_ = (const float*)d_in[3];
  const float* w_in = (const float*)d_in[4];
  const float* b_in = (const float*)d_in[5];
  const float* w_ih = (const float*)d_in[6];
  const float* w_hh = (const float*)d_in[7];
  const float* b_ih = (const float*)d_in[8];
  const float* b_hh = (const float*)d_in[9];
  const float* w_pr1 = (const float*)d_in[10];
  const float* b_pr1 = (const float*)d_in[11];
  const float* w_pr2 = (const float*)d_in[12];
  const float* b_pr2 = (const float*)d_in[13];
  const float* w_po1 = (const float*)d_in[14];
  const float* b_po1 = (const float*)d_in[15];
  const float* w_po2 = (const float*)d_in[16];
  const float* b_po2 = (const float*)d_in[17];

  bf16* ws = (bf16*)d_ws;
  size_t off = 0;
  auto take = [&](int KC, int NT) { size_t o = off; off += (size_t)KC * NT * 512; return o; };
  size_t oWin  = take(2, 13);
  size_t oWihr = take(7, 13), oWihz = take(7, 13), oWihn = take(7, 13);
  size_t oWhhr = take(7, 13), oWhhz = take(7, 13), oWhhn = take(7, 13);
  size_t oWpr1 = take(7, 13), oWpr2 = take(7, 4);
  size_t oWpo1 = take(39, 13), oWpo2 = take(7, 4);
  size_t wbytes = (off * sizeof(bf16) + 255) & ~(size_t)255;
  float* bias = (float*)((char*)d_ws + wbytes);

  auto packw = [&](const float* W, size_t o, int Kl, int N, int ld, int colOff, int rowOff,
                   int KCl, int NT, int kcTot, int kcOff) {
    int total = KCl * NT * 512;
    pack_w<<<(total + 255) / 256, 256, 0, stream>>>(W, ws + o, Kl, N, ld, colOff, rowOff,
                                                    KCl, NT, kcTot, kcOff);
  };
  packw(w_in, oWin, 36, 200, 200, 0, 0, 2, 13, 2, 0);
  packw(w_ih, oWihr, 200, 200, 600, 0, 0, 7, 13, 7, 0);
  packw(w_ih, oWihz, 200, 200, 600, 200, 0, 7, 13, 7, 0);
  packw(w_ih, oWihn, 200, 200, 600, 400, 0, 7, 13, 7, 0);
  packw(w_hh, oWhhr, 200, 200, 600, 0, 0, 7, 13, 7, 0);
  packw(w_hh, oWhhz, 200, 200, 600, 200, 0, 7, 13, 7, 0);
  packw(w_hh, oWhhn, 200, 200, 600, 400, 0, 7, 13, 7, 0);
  packw(w_pr1, oWpr1, 200, 200, 200, 0, 0, 7, 13, 7, 0);
  packw(w_pr2, oWpr2, 200, 60, 60, 0, 0, 7, 4, 7, 0);
  packw(w_po1, oWpo1, 200, 200, 200, 0, 0, 7, 13, 39, 0);     // deter rows -> kc 0..6
  packw(w_po1, oWpo1, 1024, 200, 200, 0, 200, 32, 13, 39, 7); // embed rows -> kc 7..38
  packw(w_po2, oWpo2, 200, 60, 60, 0, 0, 7, 4, 7, 0);

  pack_b<<<1, 256, 0, stream>>>(b_in, nullptr, bias + 0, 200, 0, 0, 208);
  pack_b<<<1, 256, 0, stream>>>(b_ih, b_hh, bias + 208, 200, 0, 0, 208);      // r fused
  pack_b<<<1, 256, 0, stream>>>(b_ih, b_hh, bias + 416, 200, 200, 200, 208);  // z fused
  pack_b<<<1, 256, 0, stream>>>(b_ih, nullptr, bias + 624, 200, 400, 0, 208); // xn
  pack_b<<<1, 256, 0, stream>>>(b_hh, nullptr, bias + 832, 200, 400, 0, 208); // hn
  pack_b<<<1, 256, 0, stream>>>(b_pr1, nullptr, bias + 1040, 200, 0, 0, 208);
  pack_b<<<1, 256, 0, stream>>>(b_pr2, nullptr, bias + 1248, 60, 0, 0, 64);
  pack_b<<<1, 256, 0, stream>>>(b_po1, nullptr, bias + 1312, 200, 0, 0, 208);
  pack_b<<<1, 256, 0, stream>>>(b_po2, nullptr, bias + 1520, 60, 0, 0, 64);

  rssm_kernel<<<B_ / 16, 128, 0, stream>>>(
      embed, action, np_, nq_,
      ws + oWin, ws + oWihr, ws + oWihz, ws + oWihn,
      ws + oWhhr, ws + oWhhz, ws + oWhhn,
      ws + oWpr1, ws + oWpr2, ws + oWpo1, ws + oWpo2,
      bias, (float*)d_out);
}
